// GNNEncoder_14534169329850
// MI455X (gfx1250) — matrix operand order repaired and checked by reading, not compiled, not run on hardware
//
#include <hip/hip_runtime.h>
#include <hip/hip_bf16.h>

typedef __bf16 bf16;
typedef __attribute__((ext_vector_type(8)))  __bf16 bf16x8;
typedef __attribute__((ext_vector_type(16))) __bf16 v16bf;
typedef __attribute__((ext_vector_type(8)))  float  v8f;

#define NNODES  4000
#define NEDGES  16000
#define NGRAPHS 128
#define OUTD    128

// ---------------------------------------------------------------------------
// Convert + transpose: src [K, C] f32  ->  dst [C, K] bf16
// ---------------------------------------------------------------------------
__global__ void k_cvt_transpose(const float* __restrict__ src, bf16* __restrict__ dst,
                                int K, int C) {
  long long total = (long long)K * C;
  for (long long idx = blockIdx.x * (long long)blockDim.x + threadIdx.x; idx < total;
       idx += (long long)gridDim.x * blockDim.x) {
    int c = (int)(idx / K);
    int k = (int)(idx - (long long)c * K);
    dst[(long long)c * K + k] = (bf16)src[(long long)k * C + c];
  }
}

// ---------------------------------------------------------------------------
// Node transform: out[n,o] = bias[o] + sum_i x[n,i] * root[i,o]
// ---------------------------------------------------------------------------
__global__ void k_node_transform(const float* __restrict__ x, const float* __restrict__ root,
                                 const float* __restrict__ bias, float* __restrict__ out,
                                 int n, int din, int dout) {
  int idx = blockIdx.x * blockDim.x + threadIdx.x;
  if (idx >= n * dout) return;
  int node = idx / dout, o = idx - node * dout;
  float s = bias[o];
  for (int i = 0; i < din; ++i) s += x[node * din + i] * root[i * dout + o];
  out[idx] = s;
}

__global__ void k_relu(float* __restrict__ p, int n) {
  int i = blockIdx.x * blockDim.x + threadIdx.x;
  if (i < n) p[i] = fmaxf(p[i], 0.f);
}

__global__ void k_zero(float* __restrict__ p, int n) {
  int i = blockIdx.x * blockDim.x + threadIdx.x;
  if (i < n) p[i] = 0.f;
}

// ---------------------------------------------------------------------------
// Fused NNConv edge pipeline. One block = 64 edges (four 16-row M-tiles),
// 256 threads = 8 waves. DOUT=128: each wave = 1 column-tile x 4 M-tiles.
// DOUT=64: 2 wave-rows, each wave = 1 column-tile x 2 M-tiles.
// B fragments (weights, L2-resident) stream through a depth-D (8) circular
// register pipeline with cross-`i` refill: ~2D loads in flight per wave so
// L2 latency is amortized to latency/D per WMMA group.
// ---------------------------------------------------------------------------
template <int DIN, int DOUT, int H1>
__global__ void __launch_bounds__(256)
k_edge_fused(const float* __restrict__ x,
             const int*   __restrict__ edge_index,
             const float* __restrict__ edge_attr,
             const float* __restrict__ we1,
             const float* __restrict__ be1,
             const bf16*  __restrict__ weT,   // [DIN*DOUT][H1] bf16
             const float* __restrict__ be2,
             float*       __restrict__ node_out) {
  constexpr int KB    = (H1 < 1024) ? H1 : 1024;  // K-block staged in LDS
  constexpr int KB32  = KB / 32;                  // 14 or 32
  constexpr int CT    = DOUT / 16;                // column tiles (4 or 8)
  constexpr int NWCOL = (CT < 8) ? CT : 8;        // wave-columns
  constexpr int NROW  = 8 / NWCOL;                // wave-rows (2 or 1)
  constexpr int MPW   = 4 / NROW;                 // M-tiles per wave (2 or 4)
  constexpr int D     = (KB32 % 8 == 0) ? 8 : 2;  // B prefetch pipeline depth
  constexpr int NG    = KB32 / D;                 // outer K-groups (4 or 7)

  __shared__ bf16  h_lds[64 * KB];                // 128 KB max (WGP has 320 KB)
  __shared__ float xs_lds[64 * 64];
  __shared__ float ea_lds[64 * 3];
  __shared__ int   dst_lds[64];

  const int tid  = threadIdx.x;
  const int lane = tid & 31;
  const int wave = tid >> 5;
  const int e0   = blockIdx.x * 64;

  // Stage per-edge source features, edge attrs, dst indices.
  for (int t = tid; t < 64 * DIN; t += 256) {
    int m = t / DIN, i = t - m * DIN;
    int s = edge_index[e0 + m];                  // src row
    xs_lds[m * 64 + i] = x[s * DIN + i];
  }
  for (int t = tid; t < 64 * 3; t += 256) {
    int m = t / 3, j = t - m * 3;
    ea_lds[t] = edge_attr[(e0 + m) * 3 + j];
  }
  if (tid < 64) dst_lds[tid] = edge_index[NEDGES + e0 + tid];

  // WMMA fragment lane mapping (wave32).
  const int a_row  = lane & 15;
  const int a_kofs = (lane >> 4) * 16;   // A: K {0..15} | {16..31}
  const int b_col  = lane & 15;
  const int b_kofs = (lane >> 4) * 16;   // B: K {0..15} | {16..31}
  const int m_base = (lane >> 4) * 8;    // C: rows r | r+8
  const int colt   = wave % NWCOL;       // column tile of this wave
  const int mg     = wave / NWCOL;       // M-tile group of this wave
  const int colbase = colt * 16;
  const int mt0     = mg * MPW;          // first M-tile index

  const v8f vzero = {0.f, 0.f, 0.f, 0.f, 0.f, 0.f, 0.f, 0.f};
  v8f msg[MPW];
#pragma unroll
  for (int u = 0; u < MPW; ++u) msg[u] = vzero;

  union Bu { v16bf v; bf16x8 h[2]; };

  for (int kb0 = 0; kb0 < H1; kb0 += KB) {
    __syncthreads();
    // Edge-net first layer fused: h = relu(ea @ we1 + be1) -> bf16 LDS tile.
    for (int t = tid; t < 64 * KB; t += 256) {
      int m = t / KB, k = t - m * KB;
      int kk = kb0 + k;
      float s = be1[kk]
              + ea_lds[m * 3 + 0] * we1[0 * H1 + kk]
              + ea_lds[m * 3 + 1] * we1[1 * H1 + kk]
              + ea_lds[m * 3 + 2] * we1[2 * H1 + kk];
      h_lds[m * KB + k] = (bf16)fmaxf(s, 0.f);
    }
    __syncthreads();

    // B address helper: channel i, 32-wide K chunk kc32.
    const bf16* wbase = weT + (long long)(colbase + b_col) * H1 + kb0 + b_kofs;
    auto bptr = [&](int i, int kc32) -> const bf16* {
      return wbase + (long long)i * DOUT * H1 + kc32 * 32;
    };

    // Prologue: fill B pipeline with stream positions 0..D-1 (i=0).
    Bu bpipe[D];
#pragma unroll
    for (int p = 0; p < D; ++p) {
      const bf16* bp = bptr(0, p);
      bpipe[p].h[0] = *(const bf16x8*)(bp);
      bpipe[p].h[1] = *(const bf16x8*)(bp + 8);
    }

    for (int i = 0; i < DIN; ++i) {
      v8f acc[MPW];
#pragma unroll
      for (int u = 0; u < MPW; ++u) acc[u] = vzero;

      for (int g = 0; g < NG; ++g) {        // outer K-groups (not unrolled)
#pragma unroll
        for (int kcu = 0; kcu < D; ++kcu) { // pipeline slots (unrolled)
          const int kc32 = g * D + kcu;
          // A fragments for this K chunk (LDS, low latency).
          union { v16bf v; bf16x8 h[2]; } A[MPW];
#pragma unroll
          for (int u = 0; u < MPW; ++u) {
            const int row = (mt0 + u) * 16 + a_row;
            A[u].h[0] = *(const bf16x8*)&h_lds[row * KB + kc32 * 32 + a_kofs];
            A[u].h[1] = *(const bf16x8*)&h_lds[row * KB + kc32 * 32 + a_kofs + 8];
          }
          // Consume pipeline slot kcu.
#pragma unroll
          for (int u = 0; u < MPW; ++u)
            acc[u] = __builtin_amdgcn_wmma_f32_16x16x32_bf16(
                false, A[u].v, false, bpipe[kcu].v, (short)0, acc[u], false, false);
          // Refill slot from stream position + D (crosses into i+1 on last group).
          if (g + 1 < NG) {
            const bf16* bp = bptr(i, kc32 + D);
            bpipe[kcu].h[0] = *(const bf16x8*)(bp);
            bpipe[kcu].h[1] = *(const bf16x8*)(bp + 8);
          } else if (i + 1 < DIN) {
            const bf16* bp = bptr(i + 1, kcu);
            bpipe[kcu].h[0] = *(const bf16x8*)(bp);
            bpipe[kcu].h[1] = *(const bf16x8*)(bp + 8);
          }
        }
      }
      // msg += x[src, i] * W_i   (row-wise scale of accumulators)
#pragma unroll
      for (int u = 0; u < MPW; ++u)
#pragma unroll
        for (int r = 0; r < 8; ++r) {
          float xs = xs_lds[((mt0 + u) * 16 + m_base + r) * 64 + i];
          msg[u][r] += xs * acc[u][r];
        }
    }
  }

  // Epilogue: add be2 contribution, scatter-add to destination nodes.
#pragma unroll
  for (int u = 0; u < MPW; ++u) {
    int col = colbase + b_col;
#pragma unroll
    for (int r = 0; r < 8; ++r) {
      int m = (mt0 + u) * 16 + m_base + r;
      float bt = 0.f;
      for (int i = 0; i < DIN; ++i) bt += xs_lds[m * 64 + i] * be2[i * DOUT + col];
      float v = msg[u][r] + bt;
      float* addr = node_out + (long long)dst_lds[m] * DOUT + col;
      __hip_atomic_fetch_add(addr, v, __ATOMIC_RELAXED, __HIP_MEMORY_SCOPE_AGENT);
    }
  }
}

// ---------------------------------------------------------------------------
// Global mean pool
// ---------------------------------------------------------------------------
__global__ void k_pool_acc(const float* __restrict__ h, const int* __restrict__ batch,
                           float* __restrict__ sums, float* __restrict__ cnt) {
  int idx = blockIdx.x * blockDim.x + threadIdx.x;
  if (idx >= NNODES * OUTD) return;
  int node = idx / OUTD, o = idx - node * OUTD;
  int g = batch[node];
  __hip_atomic_fetch_add(&sums[g * OUTD + o], h[idx], __ATOMIC_RELAXED,
                         __HIP_MEMORY_SCOPE_AGENT);
  if (o == 0)
    __hip_atomic_fetch_add(&cnt[g], 1.f, __ATOMIC_RELAXED, __HIP_MEMORY_SCOPE_AGENT);
}

__global__ void k_pool_div(const float* __restrict__ sums, const float* __restrict__ cnt,
                           float* __restrict__ out) {
  int idx = blockIdx.x * blockDim.x + threadIdx.x;
  if (idx >= NGRAPHS * OUTD) return;
  int g = idx / OUTD;
  out[idx] = sums[idx] / fmaxf(cnt[g], 1.f);
}

// ---------------------------------------------------------------------------
extern "C" void kernel_launch(void* const* d_in, const int* in_sizes, int n_in,
                              void* d_out, int out_size, void* d_ws, size_t ws_size,
                              hipStream_t stream) {
  (void)in_sizes; (void)n_in; (void)out_size; (void)ws_size;
  const float* x     = (const float*)d_in[0];
  const int*   eidx  = (const int*)d_in[1];
  const float* eattr = (const float*)d_in[2];
  const int*   batch = (const int*)d_in[3];

  const float *we1_0 = (const float*)d_in[4],  *be1_0 = (const float*)d_in[5],
              *we2_0 = (const float*)d_in[6],  *be2_0 = (const float*)d_in[7],
              *root0 = (const float*)d_in[8],  *bias0 = (const float*)d_in[9];
  const float *we1_1 = (const float*)d_in[10], *be1_1 = (const float*)d_in[11],
              *we2_1 = (const float*)d_in[12], *be2_1 = (const float*)d_in[13],
              *root1 = (const float*)d_in[14], *bias1 = (const float*)d_in[15];
  const float *we1_2 = (const float*)d_in[16], *be1_2 = (const float*)d_in[17],
              *we2_2 = (const float*)d_in[18], *be2_2 = (const float*)d_in[19],
              *root2 = (const float*)d_in[20], *bias2 = (const float*)d_in[21];

  // Workspace carving (~101 MB, all rewritten every launch).
  char* w = (char*)d_ws;
  auto carve = [&](size_t bytes) -> void* {
    void* p = (void*)w;
    w += (bytes + 255) & ~(size_t)255;
    return p;
  };
  bf16*  weT0 = (bf16*)carve((size_t)448 * 448 * 2);
  bf16*  weT1 = (bf16*)carve((size_t)4096 * 4096 * 2);
  bf16*  weT2 = (bf16*)carve((size_t)8192 * 4096 * 2);
  float* nb1  = (float*)carve((size_t)NNODES * 64 * 4);
  float* nb2  = (float*)carve((size_t)NNODES * 64 * 4);
  float* nb3  = (float*)carve((size_t)NNODES * OUTD * 4);
  float* sums = (float*)carve((size_t)(NGRAPHS * OUTD + NGRAPHS) * 4);
  float* cnt  = sums + NGRAPHS * OUTD;

  // Pre-pack edge-net second-layer weights: f32 [H1, DIN*DOUT] -> bf16 [DIN*DOUT, H1].
  k_cvt_transpose<<<784, 256, 0, stream>>>(we2_0, weT0, 448, 448);
  k_cvt_transpose<<<65536, 256, 0, stream>>>(we2_1, weT1, 4096, 4096);
  k_cvt_transpose<<<131072, 256, 0, stream>>>(we2_2, weT2, 4096, 8192);

  const int NB64  = (NNODES * 64 + 255) / 256;
  const int NB128 = (NNODES * OUTD + 255) / 256;

  // Layer 0: [N,7] -> [N,64]
  k_node_transform<<<NB64, 256, 0, stream>>>(x, root0, bias0, nb1, NNODES, 7, 64);
  k_edge_fused<7, 64, 448><<<NEDGES / 64, 256, 0, stream>>>(
      x, eidx, eattr, we1_0, be1_0, weT0, be2_0, nb1);
  k_relu<<<NB64, 256, 0, stream>>>(nb1, NNODES * 64);

  // Layer 1: [N,64] -> [N,64]
  k_node_transform<<<NB64, 256, 0, stream>>>(nb1, root1, bias1, nb2, NNODES, 64, 64);
  k_edge_fused<64, 64, 4096><<<NEDGES / 64, 256, 0, stream>>>(
      nb1, eidx, eattr, we1_1, be1_1, weT1, be2_1, nb2);
  k_relu<<<NB64, 256, 0, stream>>>(nb2, NNODES * 64);

  // Layer 2: [N,64] -> [N,128]
  k_node_transform<<<NB128, 256, 0, stream>>>(nb2, root2, bias2, nb3, NNODES, 64, 128);
  k_edge_fused<64, 128, 4096><<<NEDGES / 64, 256, 0, stream>>>(
      nb2, eidx, eattr, we1_2, be1_2, weT2, be2_2, nb3);

  // Global mean pool.
  k_zero<<<(NGRAPHS * OUTD + NGRAPHS + 255) / 256, 256, 0, stream>>>(
      sums, NGRAPHS * OUTD + NGRAPHS);
  k_pool_acc<<<NB128, 256, 0, stream>>>(nb3, batch, sums, cnt);
  k_pool_div<<<(NGRAPHS * OUTD + 255) / 256, 256, 0, stream>>>(sums, cnt, (float*)d_out);
}
